// TransformerBlock_57973468562037
// MI455X (gfx1250) — matrix operand order repaired and checked by reading, not compiled, not run on hardware
//
#include <hip/hip_runtime.h>
#include <stdint.h>

typedef unsigned short u16;
typedef unsigned int   u32;
typedef __attribute__((ext_vector_type(16))) __bf16 v16bf;
typedef __attribute__((ext_vector_type(8)))  float  v8f;
typedef __attribute__((ext_vector_type(8)))  u32    v8u;

#define TB   2
#define TS   2048
#define TD   1024
#define TNH  16
#define TDK  64
#define TT   (TB*TS)     // 4096 tokens
#define TE   8
#define TMOEH 1024
#define TSH  2048

// ---------------- helpers ----------------
__device__ __forceinline__ u16 f2bf(float f) {
  u32 u = __float_as_uint(f);
  u32 r = (u + 0x7FFFu + ((u >> 16) & 1u)) >> 16;
  return (u16)r;
}

__device__ __forceinline__ v8f wmma_bf16(v8u a, v8u b, v8f c) {
  return __builtin_amdgcn_wmma_f32_16x16x32_bf16(
      false, __builtin_bit_cast(v16bf, a),
      false, __builtin_bit_cast(v16bf, b),
      (short)0, c, false, false);
}

// A-operand 16x32 bf16: lane r holds row r, K = 16h..16h+15
__device__ __forceinline__ v8u load_afrag(const u16* base, int ld, int lane) {
  int r = lane & 15, h = lane >> 4;
  uint4 a = *(const uint4*)(base + r * ld + h * 16);
  uint4 b = *(const uint4*)(base + r * ld + h * 16 + 8);
  v8u f;
  f[0] = a.x; f[1] = a.y; f[2] = a.z; f[3] = a.w;
  f[4] = b.x; f[5] = b.y; f[6] = b.z; f[7] = b.w;
  return f;
}

// B-operand 32x16 bf16 from row-major W[N,K] tile: lane n holds col n, K = 16h..16h+15
__device__ __forceinline__ v8u load_bfrag(const u16* base, int ld, int lane) {
  int n = lane & 15, h = lane >> 4;
  const uint4* p = (const uint4*)(base + n * ld + h * 16);
  uint4 a = p[0], b = p[1];
  v8u f;
  f[0] = a.x; f[1] = a.y; f[2] = a.z; f[3] = a.w;
  f[4] = b.x; f[5] = b.y; f[6] = b.z; f[7] = b.w;
  return f;
}

// CDNA5 async global->LDS copy (ASYNCcnt-tracked, bypasses VGPRs)
__device__ __forceinline__ void async_ld16(u16* lds, const u16* g) {
  u32 l = (u32)(uintptr_t)lds;  // flat shared addr low 32 bits == LDS byte offset
  unsigned long long a = (unsigned long long)(uintptr_t)g;
  asm volatile("global_load_async_to_lds_b128 %0, %1, off" :: "v"(l), "v"(a) : "memory");
}

__device__ __forceinline__ void wait_async0() {
#if __has_builtin(__builtin_amdgcn_s_wait_asynccnt)
  __builtin_amdgcn_s_wait_asynccnt(0);
#else
  asm volatile("s_wait_asynccnt 0x0" ::: "memory");
#endif
}

// ---------------- fp32 -> bf16 weight conversion ----------------
__global__ void cvt_bf16_k(const float* __restrict__ src, u16* __restrict__ dst, int n) {
  for (int i = blockIdx.x * blockDim.x + threadIdx.x; i < n; i += gridDim.x * blockDim.x)
    dst[i] = f2bf(src[i]);
}

// ---------------- RMSNorm -> bf16 ----------------
__global__ void __launch_bounds__(256) rmsnorm_k(const float* __restrict__ x,
                                                 const float* __restrict__ w,
                                                 u16* __restrict__ out) {
  __shared__ float red[8];
  __shared__ float rs_s;
  int t = blockIdx.x;
  const float* xr = x + (size_t)t * TD;
  float v[4];
  float ss = 0.f;
#pragma unroll
  for (int i = 0; i < 4; ++i) { v[i] = xr[threadIdx.x + i * 256]; ss += v[i] * v[i]; }
  for (int o = 16; o > 0; o >>= 1) ss += __shfl_xor(ss, o, 32);
  if ((threadIdx.x & 31) == 0) red[threadIdx.x >> 5] = ss;
  __syncthreads();
  if (threadIdx.x == 0) {
    float s = 0.f;
    for (int i = 0; i < 8; ++i) s += red[i];
    rs_s = rsqrtf(s / (float)TD + 1e-6f);
  }
  __syncthreads();
  float rs = rs_s;
#pragma unroll
  for (int i = 0; i < 4; ++i) {
    int d = threadIdx.x + i * 256;
    out[(size_t)t * TD + d] = f2bf(v[i] * rs * w[d]);
  }
}

// ---------------- Generic WMMA GEMM: C[M,N] = A[M,K](bf16) * W[N,K]^T ----------------
// Block tile 128x128, BK=32, 8 waves each 32Mx64N (8 wmma / K-step),
// double-buffered LDS fed by async global->LDS copies.
// MODE 0: outF = acc (+bias)
// MODE 2: outH = silu(other) * (acc + bias)      (bf16 out)
// MODE 3: outF = acc + other                     (residual)
// MODE 4: outF += scale[m] * (acc + bias)        (scale==nullptr -> 1.0)
template <int MODE>
__global__ void __launch_bounds__(256) gemm_bf16_k(
    const u16* __restrict__ A, const u16* __restrict__ W,
    float* __restrict__ outF, u16* __restrict__ outH,
    int M, int N, int K,
    const float* __restrict__ bias, const float* __restrict__ other,
    const float* __restrict__ scale, int scale_stride) {
  __shared__ __align__(16) u16 As[2][128 * 32];
  __shared__ __align__(16) u16 Bs[2][128 * 32];
  const int tid = threadIdx.x;
  const int lane = tid & 31;
  const int wid = tid >> 5;
  const int wm = wid >> 1, wn = wid & 1;  // 4 M-strips x 2 N-strips
  const int bM = blockIdx.y * 128;
  const int bN = blockIdx.x * 128;

  const int lrow = tid >> 1, lch = tid & 1;
  const u16* gA = A + (size_t)(bM + lrow) * K + lch * 16;
  const u16* gB = W + (size_t)(bN + lrow) * K + lch * 16;
  u16* lA = &As[0][lrow * 32 + lch * 16];
  u16* lB = &Bs[0][lrow * 32 + lch * 16];
  const int ldsBuf = 128 * 32;  // elements per buffer

  // prologue: stage first K-slice into buffer 0
  async_ld16(lA, gA);
  async_ld16(lA + 8, gA + 8);
  async_ld16(lB, gB);
  async_ld16(lB + 8, gB + 8);

  v8f acc[2][4] = {};
  int buf = 0;
  for (int kk = 0; kk < K; kk += 32, buf ^= 1) {
    wait_async0();        // own async copies into LDS complete
    __syncthreads();      // -> every wave's copies visible; prev readers done
    if (kk + 32 < K) {    // stage next slice into the other buffer
      const u16* nA = gA + kk + 32;
      const u16* nB = gB + kk + 32;
      u16* dA = lA + (buf ^ 1 ? ldsBuf : -ldsBuf) * 0 + (buf ^ 1) * ldsBuf - buf * 0;
      // (compute plain: destination buffer is (buf^1))
      dA = &As[buf ^ 1][lrow * 32 + lch * 16];
      u16* dB = &Bs[buf ^ 1][lrow * 32 + lch * 16];
      async_ld16(dA, nA);
      async_ld16(dA + 8, nA + 8);
      async_ld16(dB, nB);
      async_ld16(dB + 8, nB + 8);
    }
    v8u af0 = load_afrag(&As[buf][(wm * 32 + 0) * 32], 32, lane);
    v8u af1 = load_afrag(&As[buf][(wm * 32 + 16) * 32], 32, lane);
#pragma unroll
    for (int nt = 0; nt < 4; ++nt) {
      v8u bf = load_bfrag(&Bs[buf][(wn * 64 + nt * 16) * 32], 32, lane);
      acc[0][nt] = wmma_bf16(af0, bf, acc[0][nt]);
      acc[1][nt] = wmma_bf16(af1, bf, acc[1][nt]);
    }
  }
  const int n_loc = lane & 15, hfl = lane >> 4;
#pragma unroll
  for (int mt = 0; mt < 2; ++mt)
#pragma unroll
    for (int nt = 0; nt < 4; ++nt) {
      int col = bN + wn * 64 + nt * 16 + n_loc;
      int row0 = bM + wm * 32 + mt * 16 + hfl * 8;
      float bv = (bias != nullptr) ? bias[col] : 0.f;
#pragma unroll
      for (int g = 0; g < 8; ++g) {
        int m = row0 + g;
        size_t idx = (size_t)m * N + col;
        float v = acc[mt][nt][g];
        if (MODE == 0) {
          outF[idx] = v + bv;
        } else if (MODE == 2) {
          float o = other[idx];
          float sl = o / (1.f + __expf(-o));
          outH[idx] = f2bf(sl * (v + bv));
        } else if (MODE == 3) {
          outF[idx] = v + other[idx];
        } else {  // MODE 4
          float s = (scale != nullptr) ? scale[(size_t)m * scale_stride] : 1.f;
          outF[idx] += s * (v + bv);
        }
      }
    }
}

// ---------------- RoPE + pack to [B,NH,S,DK] bf16, V transposed to [B,NH,DK,S] ----------------
__global__ void __launch_bounds__(256) rope_pack_k(
    const float* __restrict__ qf, const float* __restrict__ kf, const float* __restrict__ vf,
    const float* __restrict__ cosb, const float* __restrict__ sinb,
    u16* __restrict__ qb, u16* __restrict__ kb, u16* __restrict__ vt) {
  int lane = threadIdx.x & 31;  // rotary pair index 0..31
  int r = blockIdx.x * 8 + (threadIdx.x >> 5);
  int b = r / (TS * TNH);
  int rem = r - b * TS * TNH;
  int s = rem / TNH;
  int h = rem - s * TNH;
  size_t src = ((size_t)(b * TS + s) * TD) + (size_t)h * TDK;
  float c = cosb[s * 32 + lane], sn = sinb[s * 32 + lane];
  float qr = qf[src + 2 * lane], qi = qf[src + 2 * lane + 1];
  float kr = kf[src + 2 * lane], ki = kf[src + 2 * lane + 1];
  size_t dst = ((size_t)(b * TNH + h) * TS + s) * TDK;
  qb[dst + 2 * lane]     = f2bf(qr * c - qi * sn);
  qb[dst + 2 * lane + 1] = f2bf(qr * sn + qi * c);
  kb[dst + 2 * lane]     = f2bf(kr * c - ki * sn);
  kb[dst + 2 * lane + 1] = f2bf(kr * sn + ki * c);
  float v0 = vf[src + 2 * lane], v1 = vf[src + 2 * lane + 1];
  size_t vbase = (size_t)(b * TNH + h) * TDK;
  vt[(vbase + 2 * lane) * TS + s]     = f2bf(v0);
  vt[(vbase + 2 * lane + 1) * TS + s] = f2bf(v1);
}

// ---------------- Flash attention: one wave = one (b,h,16-query tile) ----------------
__global__ void __launch_bounds__(256) attn_k(const u16* __restrict__ qb,
                                              const u16* __restrict__ kb,
                                              const u16* __restrict__ vt,
                                              u16* __restrict__ ao) {
  const int lane = threadIdx.x & 31;
  const int unit = blockIdx.x * 8 + (threadIdx.x >> 5);
  const int qt = unit & (TS / 16 - 1);
  const int bh = unit >> 7;              // S/16 == 128
  const int b = bh >> 4, hh = bh & 15;   // NH == 16
  const u16* qp = qb + ((size_t)bh * TS + (size_t)qt * 16) * TDK;
  const u16* kp = kb + (size_t)bh * TS * TDK;
  const u16* vp = vt + (size_t)bh * TDK * TS;

  const int n_loc = lane & 15, hfl = lane >> 4;
  const int qidx = qt * 16 + n_loc;

  // Q fragments (B operand of S^T = K * Q^T), d-chunks 0..31 and 32..63
  v8u qfrag0 = load_bfrag(qp, TDK, lane);
  v8u qfrag1 = load_bfrag(qp + 32, TDK, lane);

  v8f O[4] = {};
  float m_run = -1e30f, l_run = 0.f;
  const int kend = 16 * (qt + 1);  // causal early exit
  for (int kc = 0; kc < kend; kc += 32) {
    v8f S0 = {}, S1 = {};
    {
      v8u a00 = load_afrag(kp + (size_t)kc * TDK, TDK, lane);
      v8u a01 = load_afrag(kp + (size_t)kc * TDK + 32, TDK, lane);
      S0 = wmma_bf16(a00, qfrag0, S0);
      S0 = wmma_bf16(a01, qfrag1, S0);
      v8u a10 = load_afrag(kp + (size_t)(kc + 16) * TDK, TDK, lane);
      v8u a11 = load_afrag(kp + (size_t)(kc + 16) * TDK + 32, TDK, lane);
      S1 = wmma_bf16(a10, qfrag0, S1);
      S1 = wmma_bf16(a11, qfrag1, S1);
    }
    // scale^2 (= (1/8)^2) and causal mask (mask * scale = -1.25e8), layout [k, q]
    float part = -3e38f;
#pragma unroll
    for (int g = 0; g < 8; ++g) {
      int k0 = kc + hfl * 8 + g;
      int k1 = kc + 16 + hfl * 8 + g;
      float v0 = S0[g] * 0.015625f + (k0 > qidx ? -1.25e8f : 0.f);
      float v1 = S1[g] * 0.015625f + (k1 > qidx ? -1.25e8f : 0.f);
      S0[g] = v0; S1[g] = v1;
      part = fmaxf(part, fmaxf(v0, v1));
    }
    part = fmaxf(part, __shfl_xor(part, 16, 32));
    float m_new = fmaxf(m_run, part);
    float corr = __expf(m_run - m_new);
    float psum = 0.f;
#pragma unroll
    for (int g = 0; g < 8; ++g) {
      float p0 = __expf(S0[g] - m_new);
      float p1 = __expf(S1[g] - m_new);
      S0[g] = p0; S1[g] = p1;
      psum += p0 + p1;
    }
    psum += __shfl_xor(psum, 16, 32);
    l_run = l_run * corr + psum;
    m_run = m_new;
#pragma unroll
    for (int dt = 0; dt < 4; ++dt)
#pragma unroll
      for (int g = 0; g < 8; ++g) O[dt][g] *= corr;

    // C-layout [k,q] -> B-operand (32k x 16q) with pack + xor16 shuffles
    v8u pf;
#pragma unroll
    for (int p = 0; p < 4; ++p) {
      u32 x0 = ((u32)f2bf(S0[2 * p])) | (((u32)f2bf(S0[2 * p + 1])) << 16);
      u32 x1 = ((u32)f2bf(S1[2 * p])) | (((u32)f2bf(S1[2 * p + 1])) << 16);
      u32 y0 = (u32)__shfl_xor((int)x0, 16, 32);
      u32 y1 = (u32)__shfl_xor((int)x1, 16, 32);
      pf[p]     = hfl ? y1 : x0;
      pf[4 + p] = hfl ? x1 : y0;
    }
    // O^T[d,q] += V^T[d,k] * P[k,q]
#pragma unroll
    for (int dt = 0; dt < 4; ++dt) {
      v8u vfr = load_afrag(vp + (size_t)(dt * 16) * TS + kc, TS, lane);
      O[dt] = wmma_bf16(vfr, pf, O[dt]);
    }
  }
  float linv = 1.f / l_run;
  size_t obase = ((size_t)(b * TS + qidx)) * TD + (size_t)hh * TDK + hfl * 8;
#pragma unroll
  for (int dt = 0; dt < 4; ++dt) {
    u32 w[4];
#pragma unroll
    for (int p = 0; p < 4; ++p) {
      float a = O[dt][2 * p] * linv, c2 = O[dt][2 * p + 1] * linv;
      w[p] = ((u32)f2bf(a)) | (((u32)f2bf(c2)) << 16);
    }
    uint4 st; st.x = w[0]; st.y = w[1]; st.z = w[2]; st.w = w[3];
    *((uint4*)&ao[obase + dt * 16]) = st;
  }
}

// ---------------- Router gate: rmsnorm dot + softmax + top-2 -> wmask[T,E] ----------------
__global__ void __launch_bounds__(256) gate_k(const float* __restrict__ x1,
                                              const float* __restrict__ nw,
                                              const float* __restrict__ gw,
                                              const float* __restrict__ gb,
                                              float* __restrict__ wmask) {
  __shared__ float xw[TD];
  __shared__ float red[8];
  __shared__ float sc[TE];
  __shared__ float rs_s;
  int t = blockIdx.x;
  const float* xr = x1 + (size_t)t * TD;
  float ss = 0.f;
  for (int i = threadIdx.x; i < TD; i += 256) {
    float v = xr[i];
    ss += v * v;
    xw[i] = v * nw[i];
  }
  for (int o = 16; o > 0; o >>= 1) ss += __shfl_xor(ss, o, 32);
  if ((threadIdx.x & 31) == 0) red[threadIdx.x >> 5] = ss;
  __syncthreads();
  if (threadIdx.x == 0) {
    float s = 0.f;
    for (int i = 0; i < 8; ++i) s += red[i];
    rs_s = rsqrtf(s / (float)TD + 1e-6f);
  }
  __syncthreads();
  int e = threadIdx.x >> 5, lane = threadIdx.x & 31;
  float acc = 0.f;
  for (int d = lane; d < TD; d += 32) acc += xw[d] * gw[(size_t)e * TD + d];
  for (int o = 16; o > 0; o >>= 1) acc += __shfl_xor(acc, o, 32);
  if (lane == 0) sc[e] = acc * rs_s + gb[e];
  __syncthreads();
  if (threadIdx.x == 0) {
    float p[TE];
    float mx = -3e38f;
    for (int i = 0; i < TE; ++i) mx = fmaxf(mx, sc[i]);
    float s = 0.f;
    for (int i = 0; i < TE; ++i) { p[i] = __expf(sc[i] - mx); s += p[i]; }
    for (int i = 0; i < TE; ++i) p[i] /= s;
    int i1 = 0;
    for (int i = 1; i < TE; ++i) if (p[i] > p[i1]) i1 = i;
    int i2 = (i1 == 0) ? 1 : 0;
    for (int i = 0; i < TE; ++i) if (i != i1 && p[i] > p[i2]) i2 = i;
    for (int i = 0; i < TE; ++i)
      wmask[(size_t)t * TE + i] = (i == i1) ? p[i1] : ((i == i2) ? p[i2] : 0.f);
  }
}

// ---------------- host launcher ----------------
extern "C" void kernel_launch(void* const* d_in, const int* in_sizes, int n_in,
                              void* d_out, int out_size, void* d_ws, size_t ws_size,
                              hipStream_t stream) {
  (void)in_sizes; (void)n_in; (void)out_size; (void)ws_size;
  const float* x    = (const float*)d_in[0];
  const float* cosb = (const float*)d_in[1];
  const float* sinb = (const float*)d_in[2];
  /* d_in[3] mask: causal mask re-derived in-kernel */
  const float* wq   = (const float*)d_in[4];
  const float* wk   = (const float*)d_in[5];
  const float* wv   = (const float*)d_in[6];
  const float* wo   = (const float*)d_in[7];
  const float* anw  = (const float*)d_in[8];
  const float* fnw  = (const float*)d_in[9];
  const float* gw   = (const float*)d_in[10];
  const float* gb   = (const float*)d_in[11];
  const float* w1   = (const float*)d_in[12];
  const float* b1   = (const float*)d_in[13];
  const float* w2   = (const float*)d_in[14];
  const float* b2   = (const float*)d_in[15];
  const float* w3   = (const float*)d_in[16];
  const float* b3   = (const float*)d_in[17];
  const float* sgw  = (const float*)d_in[18];
  const float* suw  = (const float*)d_in[19];
  const float* sdw  = (const float*)d_in[20];
  float* out = (float*)d_out;

  char* base = (char*)d_ws;
  size_t off = 0;
  auto alloc = [&](size_t bytes) -> void* {
    void* p = base + off;
    off += (bytes + 255) & ~(size_t)255;
    return p;
  };

  // bf16 weights
  u16* wq_b = (u16*)alloc((size_t)TD * TD * 2);
  u16* wk_b = (u16*)alloc((size_t)TD * TD * 2);
  u16* wv_b = (u16*)alloc((size_t)TD * TD * 2);
  u16* wo_b = (u16*)alloc((size_t)TD * TD * 2);
  u16* w1_b = (u16*)alloc((size_t)TE * TMOEH * TD * 2);
  u16* w2_b = (u16*)alloc((size_t)TE * TD * TMOEH * 2);
  u16* w3_b = (u16*)alloc((size_t)TE * TMOEH * TD * 2);
  u16* sg_b = (u16*)alloc((size_t)TSH * TD * 2);
  u16* su_b = (u16*)alloc((size_t)TSH * TD * 2);
  u16* sd_b = (u16*)alloc((size_t)TD * TSH * 2);
  // activations
  u16* h_b = (u16*)alloc((size_t)TT * TD * 2);              // h, later hf
  char* big0 = (char*)alloc((size_t)TT * TD * 4 * 3);       // qf|kf|vf -> a1|sgbuf
  float* qf = (float*)big0;
  float* kf = qf + (size_t)TT * TD;
  float* vf = kf + (size_t)TT * TD;
  float* a1buf = (float*)big0;                              // T x MOE_H (16MB)
  float* sgbuf = (float*)(big0 + (size_t)TT * TD * 4);      // T x SH (32MB)
  char* big1 = (char*)alloc((size_t)TT * TD * 2 * 3);       // qb|kb|vt -> act_sh|act_e
  u16* qb2 = (u16*)big1;
  u16* kb2 = qb2 + (size_t)TT * TD;
  u16* vt2 = kb2 + (size_t)TT * TD;
  u16* act_sh = (u16*)big1;                                 // T x SH bf16 (16MB)
  u16* act_e = (u16*)(big1 + (size_t)TT * TSH * 2);         // T x MOE_H bf16 (8MB)
  u16* ao_b = (u16*)alloc((size_t)TT * TD * 2);
  float* wmaskb = (float*)alloc((size_t)TT * TE * 4);

  // 1) weights -> bf16
  cvt_bf16_k<<<2048, 256, 0, stream>>>(wq, wq_b, TD * TD);
  cvt_bf16_k<<<2048, 256, 0, stream>>>(wk, wk_b, TD * TD);
  cvt_bf16_k<<<2048, 256, 0, stream>>>(wv, wv_b, TD * TD);
  cvt_bf16_k<<<2048, 256, 0, stream>>>(wo, wo_b, TD * TD);
  cvt_bf16_k<<<4096, 256, 0, stream>>>(w1, w1_b, TE * TMOEH * TD);
  cvt_bf16_k<<<4096, 256, 0, stream>>>(w2, w2_b, TE * TD * TMOEH);
  cvt_bf16_k<<<4096, 256, 0, stream>>>(w3, w3_b, TE * TMOEH * TD);
  cvt_bf16_k<<<4096, 256, 0, stream>>>(sgw, sg_b, TSH * TD);
  cvt_bf16_k<<<4096, 256, 0, stream>>>(suw, su_b, TSH * TD);
  cvt_bf16_k<<<4096, 256, 0, stream>>>(sdw, sd_b, TD * TSH);

  // 2) attn rmsnorm
  rmsnorm_k<<<TT, 256, 0, stream>>>(x, anw, h_b);

  // 3) QKV projections
  dim3 gNN(TD / 128, TT / 128);
  gemm_bf16_k<0><<<gNN, 256, 0, stream>>>(h_b, wq_b, qf, nullptr, TT, TD, TD, nullptr, nullptr, nullptr, 0);
  gemm_bf16_k<0><<<gNN, 256, 0, stream>>>(h_b, wk_b, kf, nullptr, TT, TD, TD, nullptr, nullptr, nullptr, 0);
  gemm_bf16_k<0><<<gNN, 256, 0, stream>>>(h_b, wv_b, vf, nullptr, TT, TD, TD, nullptr, nullptr, nullptr, 0);

  // 4) RoPE + head-major pack (V transposed)
  rope_pack_k<<<(TB * TS * TNH) / 8, 256, 0, stream>>>(qf, kf, vf, cosb, sinb, qb2, kb2, vt2);

  // 5) causal flash attention
  attn_k<<<(TB * TNH * (TS / 16)) / 8, 256, 0, stream>>>(qb2, kb2, vt2, ao_b);

  // 6) output projection + residual -> d_out (x1)
  gemm_bf16_k<3><<<gNN, 256, 0, stream>>>(ao_b, wo_b, out, nullptr, TT, TD, TD, nullptr, x, nullptr, 0);

  // 7) ffn rmsnorm -> hf (reuse h_b)
  rmsnorm_k<<<TT, 256, 0, stream>>>(out, fnw, h_b);

  // 8) router gate (fp32 path) -> wmask
  gate_k<<<TT, 256, 0, stream>>>(out, fnw, gw, gb, wmaskb);

  // 9) shared expert: silu(hf@sg^T) * (hf@su^T) @ sd^T, accumulated into d_out
  dim3 gSH(TSH / 128, TT / 128);
  gemm_bf16_k<0><<<gSH, 256, 0, stream>>>(h_b, sg_b, sgbuf, nullptr, TT, TSH, TD, nullptr, nullptr, nullptr, 0);
  gemm_bf16_k<2><<<gSH, 256, 0, stream>>>(h_b, su_b, nullptr, act_sh, TT, TSH, TD, nullptr, sgbuf, nullptr, 0);
  gemm_bf16_k<4><<<gNN, 256, 0, stream>>>(act_sh, sd_b, out, nullptr, TT, TD, TSH, nullptr, nullptr, nullptr, 0);

  // 10) MoE experts (dense, masked accumulate per reference semantics)
  dim3 gH(TMOEH / 128, TT / 128);
  for (int e = 0; e < TE; ++e) {
    const u16* w1e = w1_b + (size_t)e * TMOEH * TD;
    const u16* w3e = w3_b + (size_t)e * TMOEH * TD;
    const u16* w2e = w2_b + (size_t)e * TD * TMOEH;
    gemm_bf16_k<0><<<gH, 256, 0, stream>>>(h_b, w1e, a1buf, nullptr, TT, TMOEH, TD,
                                           b1 + (size_t)e * TMOEH, nullptr, nullptr, 0);
    gemm_bf16_k<2><<<gH, 256, 0, stream>>>(h_b, w3e, nullptr, act_e, TT, TMOEH, TD,
                                           b3 + (size_t)e * TMOEH, a1buf, nullptr, 0);
    gemm_bf16_k<4><<<gNN, 256, 0, stream>>>(act_e, w2e, out, nullptr, TT, TD, TMOEH,
                                            b2 + (size_t)e * TD, wmaskb + e, wmaskb + e, TE);
  }
}
